// Attention_1752346656894
// MI455X (gfx1250) — matrix operand order repaired and checked by reading, not compiled, not run on hardware
//
#include <hip/hip_runtime.h>
#include <hip/hip_bf16.h>

#define DIM       384
#define THREE_DIM 1152
#define HEADS     12
#define HD        32
#define NTOK      512
#define NWIN      64
#define NG        8
#define POSL      3375      // 15*15*15
#define POSD      24
#define SCALE     0.17677669529663687f   // 1/sqrt(32)

typedef __attribute__((ext_vector_type(16))) _Float16     v16h;
typedef __attribute__((ext_vector_type(8)))  float        v8f;
typedef __attribute__((ext_vector_type(4)))  unsigned int u32x4;
typedef __attribute__((ext_vector_type(8)))  int          i32x8;
typedef __attribute__((ext_vector_type(4)))  int          i32x4;

static __device__ __forceinline__ v8f wmma_f16(v16h a, v16h b, v8f c) {
  // D = A(16x32,f16) * B(32x16,f16) + C(16x16,f32)
  return __builtin_amdgcn_wmma_f32_16x16x32_f16(false, a, false, b, (short)0, c,
                                                false, false);
}

// ---------------------------------------------------------------------------
// TDM: DMA a 2D f16 tile (tile_d0 elems wide x tile_d1 rows, row stride
// stride0 elems) from global memory into LDS at byte offset lds_off.
// Descriptor bitfields per CDNA5 ISA.
// This toolchain exposes the 6-arg builtin:
//   (u32x4 g0, i32x8 g1, i32x4 g2, i32x4 g3, i32x8 extra, i32 cpol)
// ---------------------------------------------------------------------------
static __device__ __forceinline__ void tdm_load_2d_f16(
    unsigned lds_off, const void* gptr,
    unsigned tile_d0, unsigned tile_d1,
    unsigned tensor_d0, unsigned tensor_d1,
    unsigned long long stride0_elems) {
  unsigned long long ga = (unsigned long long)gptr;
  u32x4 g0;
  g0[0] = 1u;                                   // count=1, user mode, no gather
  g0[1] = lds_off;                              // [63:32]  lds_addr (bytes)
  g0[2] = (unsigned)(ga & 0xffffffffu);         // [95:64]  global_addr[31:0]
  g0[3] = (unsigned)((ga >> 32) & 0x01ffffffu)  // [120:96] global_addr[56:32]
          | (2u << 30);                         // [127:126] type = 2 ("image")
  i32x8 g1;
  g1[0] = (int)(1u << 16);                      // wg_mask=0 (no cluster), data_size=1 (2B)
  g1[1] = (int)((tensor_d0 & 0xffffu) << 16);   // [63:48] tensor_dim0 lo16
  g1[2] = (int)(((tensor_d0 >> 16) & 0xffffu)   // [79:64] tensor_dim0 hi16
          | ((tensor_d1 & 0xffffu) << 16));     // [95:80] tensor_dim1 lo16
  g1[3] = (int)(((tensor_d1 >> 16) & 0xffffu)   // [111:96] tensor_dim1 hi16
          | ((tile_d0 & 0xffffu) << 16));       // [127:112] tile_dim0
  g1[4] = (int)(tile_d1 & 0xffffu);             // [143:128] tile_dim1, tile_dim2=0
  g1[5] = (int)(stride0_elems & 0xffffffffu);   // [191:160] dim0_stride lo32
  g1[6] = (int)((stride0_elems >> 32) & 0xffffu); // [207:192] dim0_stride hi16
  g1[7] = 0;                                    // dim1_stride unused (2D)
  i32x4 z4 = {0, 0, 0, 0};
  i32x8 z8 = {0, 0, 0, 0, 0, 0, 0, 0};
  __builtin_amdgcn_tensor_load_to_lds(g0, g1, z4, z4, z8, 0);
}

// ---------------------------------------------------------------------------
// Dynamic position-bias MLP: table[L=3375][12]
// ---------------------------------------------------------------------------
static __device__ __forceinline__ void ln_relu(const float* __restrict__ g,
                                               const float* __restrict__ b,
                                               const float* __restrict__ t,
                                               float* __restrict__ u) {
  float m = 0.f;
  #pragma unroll
  for (int j = 0; j < POSD; ++j) m += t[j];
  m *= (1.0f / POSD);
  float v = 0.f;
  #pragma unroll
  for (int j = 0; j < POSD; ++j) { float d = t[j] - m; v += d * d; }
  v *= (1.0f / POSD);
  float inv = rsqrtf(v + 1e-5f);
  #pragma unroll
  for (int j = 0; j < POSD; ++j) {
    float y = (t[j] - m) * inv * g[j] + b[j];
    u[j] = y > 0.f ? y : 0.f;
  }
}

__global__ __launch_bounds__(128) void pos_mlp_kernel(
    const float* __restrict__ pw,  const float* __restrict__ pb,
    const float* __restrict__ g1,  const float* __restrict__ b1,
    const float* __restrict__ w1,  const float* __restrict__ wb1,
    const float* __restrict__ g2,  const float* __restrict__ b2,
    const float* __restrict__ w2,  const float* __restrict__ wb2,
    const float* __restrict__ g3,  const float* __restrict__ b3,
    const float* __restrict__ w3,  const float* __restrict__ wb3,
    float* __restrict__ table) {
  int l = blockIdx.x * 128 + threadIdx.x;
  if (l >= POSL) return;
  float c0 = (float)(l / 225)       - 7.f;
  float c1 = (float)((l / 15) % 15) - 7.f;
  float c2 = (float)(l % 15)        - 7.f;

  float t[POSD], u[POSD], t2[POSD];
  #pragma unroll
  for (int j = 0; j < POSD; ++j)
    t[j] = c0 * pw[j] + c1 * pw[POSD + j] + c2 * pw[2 * POSD + j] + pb[j];

  ln_relu(g1, b1, t, u);
  for (int j = 0; j < POSD; ++j) {
    float s = wb1[j];
    for (int i = 0; i < POSD; ++i) s += u[i] * w1[i * POSD + j];
    t2[j] = s;
  }
  ln_relu(g2, b2, t2, u);
  for (int j = 0; j < POSD; ++j) {
    float s = wb2[j];
    for (int i = 0; i < POSD; ++i) s += u[i] * w2[i * POSD + j];
    t[j] = s;
  }
  ln_relu(g3, b3, t, u);
  for (int h = 0; h < HEADS; ++h) {
    float s = wb3[h];
    for (int i = 0; i < POSD; ++i) s += u[i] * w3[i * HEADS + h];
    table[l * HEADS + h] = s;
  }
}

// ---------------------------------------------------------------------------
// QKV GEMM: (32768 x 384) f32 @ (384 x 1152) f32 -> f16, q pre-scaled
// ---------------------------------------------------------------------------
__global__ __launch_bounds__(128) void qkv_gemm_kernel(
    const float* __restrict__ x, const float* __restrict__ w,
    const float* __restrict__ bias, _Float16* __restrict__ out) {
  const int NT = THREE_DIM / 16;  // 72 tiles along N
  int wave = blockIdx.x * 4 + (threadIdx.x >> 5);
  int lane = threadIdx.x & 31;
  int tm = wave / NT, tn = wave % NT;
  int r16 = lane & 15;
  int hi  = lane >> 4;
  int ak  = hi ? 8 : 0;    // unused; A fragment also uses bk (k = 16*hi + j)
  int bk  = hi ? 16 : 0;   // B-fragment K base (32x16 f16 layout)

  const float* arow = x + (size_t)(tm * 16 + r16) * DIM;
  const float* bcol = w + (size_t)bk * THREE_DIM + tn * 16 + r16;

  v8f acc = {};
  for (int kk = 0; kk < DIM; kk += 32) {
    __builtin_prefetch(arow + kk + 32, 0, 0);
    v16h a, b;
    #pragma unroll
    for (int j = 0; j < 8; ++j) {
      a[j]     = (_Float16)arow[kk + bk + j];
      a[8 + j] = (_Float16)arow[kk + bk + 8 + j];
    }
    #pragma unroll
    for (int j = 0; j < 16; ++j) b[j] = (_Float16)bcol[(size_t)(kk + j) * THREE_DIM];
    acc = wmma_f16(a, b, acc);
  }

  int col = tn * 16 + r16;
  float sc = (col < DIM) ? SCALE : 1.0f;   // scale only the q block
  float bb = bias[col];
  #pragma unroll
  for (int r = 0; r < 8; ++r) {
    int m = tm * 16 + r + hi * 8;
    out[(size_t)m * THREE_DIM + col] = (_Float16)((acc[r] + bb) * sc);
  }
}

// ---------------------------------------------------------------------------
// Fused flash attention: one wave per (16-row q tile, head, window).
// V tiles are double-buffered into LDS by the Tensor Data Mover so the DMA
// of chunk c+1 overlaps the WMMA + softmax of chunk c.
// ---------------------------------------------------------------------------
__global__ __launch_bounds__(32) void attn_kernel(
    const _Float16* __restrict__ qkvh, const float* __restrict__ pos_table,
    const float* __restrict__ mask, _Float16* __restrict__ outh) {
  __shared__ _Float16 v_lds[2][32 * 32];  // double-buffered 32-key x 32-d V tile
  __shared__ _Float16 p_lds[16 * 32];     // P staging (C-layout -> A-layout)

  int lane = threadIdx.x & 31;
  int qt   = blockIdx.x;   // 0..31
  int head = blockIdx.y;   // 0..11
  int win  = blockIdx.z;   // 0..63
  int g    = win & (NG - 1);
  int r16  = lane & 15;
  int hi   = lane >> 4;
  int ak   = hi ? 8 : 0;
  int bk   = hi ? 16 : 0;
  int qrow = win * NTOK + qt * 16;

  unsigned vbuf_off[2] = {(unsigned)(size_t)&v_lds[0][0],
                          (unsigned)(size_t)&v_lds[1][0]};
  const _Float16* vgbase =
      qkvh + (size_t)win * NTOK * THREE_DIM + 2 * DIM + head * HD;

  // Q fragment (16x32 f16, d = K dim), q already scaled
  v16h qa;
  {
    const _Float16* qp = qkvh + (size_t)(qrow + r16) * THREE_DIM + head * HD;
    #pragma unroll
    for (int j = 0; j < 8; ++j) { qa[j] = qp[bk + j]; qa[8 + j] = qp[bk + 8 + j]; }
  }

  // TDM prologue: DMA V tile of chunk 0 into buffer 0
  tdm_load_2d_f16(vbuf_off[0], vgbase, HD, 32, HD, 32, THREE_DIM);

  v8f o0 = {}, o1 = {};
  float mrun[8], lrun[8];
  #pragma unroll
  for (int r = 0; r < 8; ++r) { mrun[r] = -1e30f; lrun[r] = 0.f; }

  int p = 0;
  for (int c = 0; c < NTOK; c += 32, p ^= 1) {
    // wait for this chunk's V DMA; then kick off the next chunk's DMA into
    // the other buffer so it runs under the compute below
    __builtin_amdgcn_s_wait_tensorcnt(0);
    if (c + 32 < NTOK)
      tdm_load_2d_f16(vbuf_off[p ^ 1],
                      vgbase + (size_t)(c + 32) * THREE_DIM,
                      HD, 32, HD, 32, THREE_DIM);

    // ---- S = Q * K^T for 32 keys (two 16x16 WMMAs; contiguous per-lane
    //      16-half reads from each key's d-vector) ----
    v16h kb0, kb1;
    {
      const _Float16* kp =
          qkvh + (size_t)(win * NTOK + c + r16) * THREE_DIM + DIM + head * HD + bk;
      #pragma unroll
      for (int j = 0; j < 16; ++j) kb0[j] = kp[j];
      kp += (size_t)16 * THREE_DIM;
      #pragma unroll
      for (int j = 0; j < 16; ++j) kb1[j] = kp[j];
    }
    v8f z = {};
    v8f s0 = wmma_f16(qa, kb0, z);
    v8f s1 = wmma_f16(qa, kb1, z);

    // ---- add dynamic relative-position bias + attention mask ----
    #pragma unroll
    for (int r = 0; r < 8; ++r) {
      int qtok = qt * 16 + r + hi * 8;
      int qz = qtok >> 6, qy = (qtok >> 3) & 7, qx = qtok & 7;
      int kt0 = c + r16, kt1 = c + 16 + r16;
      int i0 = (((qz - (kt0 >> 6)) + 7) * 15 + ((qy - ((kt0 >> 3) & 7)) + 7)) * 15 +
               ((qx - (kt0 & 7)) + 7);
      int i1 = (((qz - (kt1 >> 6)) + 7) * 15 + ((qy - ((kt1 >> 3) & 7)) + 7)) * 15 +
               ((qx - (kt1 & 7)) + 7);
      s0[r] += pos_table[i0 * HEADS + head] +
               mask[((size_t)g * NTOK + qtok) * NTOK + kt0];
      s1[r] += pos_table[i1 * HEADS + head] +
               mask[((size_t)g * NTOK + qtok) * NTOK + kt1];
    }

    // ---- online softmax; row stats reduced over each 16-lane half ----
    #pragma unroll
    for (int r = 0; r < 8; ++r) {
      float rv = fmaxf(s0[r], s1[r]);
      #pragma unroll
      for (int off = 1; off < 16; off <<= 1) rv = fmaxf(rv, __shfl_xor(rv, off, 32));
      float mn    = fmaxf(mrun[r], rv);
      float alpha = __expf(mrun[r] - mn);
      float p0 = __expf(s0[r] - mn), p1 = __expf(s1[r] - mn);
      float rs = p0 + p1;
      #pragma unroll
      for (int off = 1; off < 16; off <<= 1) rs += __shfl_xor(rs, off, 32);
      lrun[r] = lrun[r] * alpha + rs;
      mrun[r] = mn;
      o0[r] *= alpha;
      o1[r] *= alpha;
      int m = r + hi * 8;                       // C-layout row
      p_lds[m * 32 + r16]      = (_Float16)p0;  // stage P for A-layout reload
      p_lds[m * 32 + 16 + r16] = (_Float16)p1;
    }
    __syncthreads();

    // ---- reload P as A-fragment (16x32) from LDS ----
    v16h pa;
    {
      const _Float16* pp = p_lds + r16 * 32;
      #pragma unroll
      for (int j = 0; j < 8; ++j) { pa[j] = pp[bk + j]; pa[8 + j] = pp[bk + 8 + j]; }
    }
    __syncthreads();

    // ---- O += P * V: V B-fragments read from the TDM-filled LDS tile ----
    const _Float16* vl = &v_lds[p][0];
    v16h vb0, vb1;
    #pragma unroll
    for (int j = 0; j < 16; ++j) vb0[j] = vl[(bk + j) * HD + r16];
    #pragma unroll
    for (int j = 0; j < 16; ++j) vb1[j] = vl[(bk + j) * HD + 16 + r16];
    o0 = wmma_f16(pa, vb0, o0);
    o1 = wmma_f16(pa, vb1, o1);
  }

  #pragma unroll
  for (int r = 0; r < 8; ++r) {
    float inv = 1.0f / lrun[r];
    int m = qrow + r + hi * 8;
    outh[(size_t)m * DIM + head * HD + r16]      = (_Float16)(o0[r] * inv);
    outh[(size_t)m * DIM + head * HD + 16 + r16] = (_Float16)(o1[r] * inv);
  }
}

// ---------------------------------------------------------------------------
// Output projection: (32768 x 384) f16 @ (384 x 384) f32 -> f32 + bias
// ---------------------------------------------------------------------------
__global__ __launch_bounds__(128) void proj_gemm_kernel(
    const _Float16* __restrict__ a, const float* __restrict__ w,
    const float* __restrict__ bias, float* __restrict__ out) {
  const int NT = DIM / 16;  // 24
  int wave = blockIdx.x * 4 + (threadIdx.x >> 5);
  int lane = threadIdx.x & 31;
  int tm = wave / NT, tn = wave % NT;
  int r16 = lane & 15;
  int hi  = lane >> 4;
  int ak  = hi ? 8 : 0;
  int bk  = hi ? 16 : 0;

  const _Float16* arow = a + (size_t)(tm * 16 + r16) * DIM;
  const float*    bcol = w + (size_t)bk * DIM + tn * 16 + r16;

  v8f acc = {};
  for (int kk = 0; kk < DIM; kk += 32) {
    __builtin_prefetch(arow + kk + 32, 0, 0);
    v16h av, b;
    #pragma unroll
    for (int j = 0; j < 8; ++j) {
      av[j]     = arow[kk + bk + j];
      av[8 + j] = arow[kk + bk + 8 + j];
    }
    #pragma unroll
    for (int j = 0; j < 16; ++j) b[j] = (_Float16)bcol[(size_t)(kk + j) * DIM];
    acc = wmma_f16(av, b, acc);
  }

  int col = tn * 16 + r16;
  float bb = bias[col];
  #pragma unroll
  for (int r = 0; r < 8; ++r) {
    int m = tm * 16 + r + hi * 8;
    out[(size_t)m * DIM + col] = acc[r] + bb;
  }
}

// ---------------------------------------------------------------------------
extern "C" void kernel_launch(void* const* d_in, const int* in_sizes, int n_in,
                              void* d_out, int out_size, void* d_ws, size_t ws_size,
                              hipStream_t stream) {
  (void)in_sizes; (void)n_in; (void)out_size; (void)ws_size;

  const float* x      = (const float*)d_in[0];
  // d_in[1..3] = H, W, D scalars (compile-time constants here)
  const float* mask   = (const float*)d_in[4];
  const float* qkv_w  = (const float*)d_in[5];
  const float* qkv_b  = (const float*)d_in[6];
  const float* proj_w = (const float*)d_in[7];
  const float* proj_b = (const float*)d_in[8];
  const float* ppw    = (const float*)d_in[9];
  const float* ppb    = (const float*)d_in[10];
  const float* g1  = (const float*)d_in[11]; const float* b1  = (const float*)d_in[12];
  const float* w1  = (const float*)d_in[13]; const float* wb1 = (const float*)d_in[14];
  const float* g2  = (const float*)d_in[15]; const float* b2  = (const float*)d_in[16];
  const float* w2  = (const float*)d_in[17]; const float* wb2 = (const float*)d_in[18];
  const float* g3  = (const float*)d_in[19]; const float* b3  = (const float*)d_in[20];
  const float* w3  = (const float*)d_in[21]; const float* wb3 = (const float*)d_in[22];

  // workspace layout:
  //   [0, 160KB)              pos_table  (3375*12 f32)
  //   [160KB, +75'497'472)    qkv in f16 (32768 x 1152)
  //   [..,  +25'165'824)      attention output in f16 (32768 x 384)
  char* ws = (char*)d_ws;
  float*    pos_table = (float*)ws;
  _Float16* qkvh = (_Float16*)(ws + (160 << 10));
  _Float16* aoh  = (_Float16*)(ws + (160 << 10) +
                               (size_t)NWIN * NTOK * THREE_DIM * sizeof(_Float16));

  pos_mlp_kernel<<<dim3((POSL + 127) / 128), dim3(128), 0, stream>>>(
      ppw, ppb, g1, b1, w1, wb1, g2, b2, w2, wb2, g3, b3, w3, wb3, pos_table);

  const int MT = (NWIN * NTOK) / 16;  // 2048 row tiles
  qkv_gemm_kernel<<<dim3(MT * (THREE_DIM / 16) / 4), dim3(128), 0, stream>>>(
      x, qkv_w, qkv_b, qkvh);

  attn_kernel<<<dim3(NTOK / 16, HEADS, NWIN), dim3(32), 0, stream>>>(
      qkvh, pos_table, mask, aoh);

  proj_gemm_kernel<<<dim3(MT * (DIM / 16) / 4), dim3(128), 0, stream>>>(
      aoh, proj_w, proj_b, (float*)d_out);
}
